// VaryMambaModel_6914897346922
// MI455X (gfx1250) — matrix operand order repaired and checked by reading, not compiled, not run on hardware
//
#include <hip/hip_runtime.h>
#include <hip/hip_bf16.h>
#include <math.h>

typedef _Float16 f16;
typedef __attribute__((ext_vector_type(16))) _Float16 v16h;
typedef __attribute__((ext_vector_type(4)))  _Float16 v4h;
typedef __attribute__((ext_vector_type(8)))  float    v8f;
typedef __attribute__((ext_vector_type(4)))  float    v4f;

// ---------------------------------------------------------------------------
// Problem constants (from the reference)
// ---------------------------------------------------------------------------
#define BATCH 2
#define SEQL  256
#define MROWS (BATCH * SEQL)   // 512
#define NSTATE 64
#define KCONV  4

// GEMM tiling: block tile 64(M) x 128(N) x 32(K); 256 threads = 8 waves
// arranged 4(M) x 2(N); each wave owns a 16x64 output strip = 4 WMMA accums.
#define BM 64
#define BN 128
#define KT 32
#define LDPAD 8

static __device__ inline v8f wmma_f16(v16h a, v16h b, v8f c) {
    // D = A(16x32 f16) * B(32x16 f16) + C(16x16 f32)
    return __builtin_amdgcn_wmma_f32_16x16x32_f16(
        /*neg_a=*/false, a, /*neg_b=*/false, b,
        /*c_mod=*/(short)0, c, /*reuse_a=*/false, /*reuse_b=*/false);
}

// ---------------------------------------------------------------------------
// GEMM: C[512,N] (f32) = A[512,K] (f32, stride lda) * W[K,N] (f32, stride ldb)
// Requirements: K % 32 == 0, N % 4 == 0, grid.y * BM == 512 (no M guards).
// f32 -> f16 conversion happens in-register on the way into LDS.
// ---------------------------------------------------------------------------
__global__ __launch_bounds__(256)
void gemm_wmma_f16_kernel(const float* __restrict__ A, int lda,
                          const float* __restrict__ W, int ldb,
                          float* __restrict__ C, int ldc,
                          int N, int Kdim)
{
    __shared__ f16 As[BM][KT + LDPAD];     // [m][k]
    __shared__ f16 Bs[BN][KT + LDPAD];     // [n][k]  (transposed store)

    const int tid  = threadIdx.x;
    const int lane = tid & 31;
    const int wave = tid >> 5;
    const int wm   = wave & 3;     // wave M index (0..3) -> 16 rows each
    const int wn   = wave >> 2;    // wave N index (0..1) -> 64 cols each
    const int m0   = blockIdx.y * BM;
    const int n0   = blockIdx.x * BN;

    const int hsel = lane >> 4;    // lane half select (0 | 1)
    const int l16  = lane & 15;

    v8f acc[4] = {};               // 16x64 strip = 4 x (16x16) f32 accumulators

    const v4f vzero = {0.f, 0.f, 0.f, 0.f};

    for (int k0 = 0; k0 < Kdim; k0 += KT) {
        // ---- phase 1: issue all global loads (no exec-mask predication) ----
        // A tile: 64x32 = 512 float4, 2 per thread (always in-bounds)
        v4f av[2];
        #pragma unroll
        for (int i = 0; i < 2; ++i) {
            int idx = tid + i * 256;           // 0..511
            int r   = idx >> 3;                // row (8 float4 per row)
            int c   = (idx & 7) << 2;          // k
            av[i] = *(const v4f*)&A[(size_t)(m0 + r) * lda + (k0 + c)];
        }
        // B tile: 32x128 = 1024 float4, 4 per thread; clamp addr, select value
        v4f bv[4];
        int  bc[4];
        bool bok[4];
        #pragma unroll
        for (int i = 0; i < 4; ++i) {
            int idx = tid + i * 256;           // 0..1023
            int r   = idx >> 5;                // k row (32 float4 per row)
            int c   = (idx & 31) << 2;         // n col in tile
            int gn  = n0 + c;
            bc[i]   = c;
            bok[i]  = (gn < N);                // N % 4 == 0 -> float4 all-or-none
            int gsafe = bok[i] ? gn : 0;
            v4f v = *(const v4f*)&W[(size_t)(k0 + r) * ldb + gsafe];
            bv[i] = bok[i] ? v : vzero;        // cndmask, no branch
        }
        // prefetch next K-slab of W into L2 (global_prefetch_b8)
        if (k0 + KT < Kdim) {
            const float* p = &W[(size_t)(k0 + KT + (tid >> 4)) * ldb + n0 + ((tid & 15) << 3)];
            __builtin_prefetch(p, 0, 1);
        }

        // ---- phase 2: convert + store tiles to LDS ----
        #pragma unroll
        for (int i = 0; i < 2; ++i) {
            int idx = tid + i * 256;
            int r   = idx >> 3;
            int c   = (idx & 7) << 2;
            v4h p;
            p[0] = (f16)av[i][0]; p[1] = (f16)av[i][1];
            p[2] = (f16)av[i][2]; p[3] = (f16)av[i][3];
            *(v4h*)&As[r][c] = p;              // ds_store_b64
        }
        #pragma unroll
        for (int i = 0; i < 4; ++i) {
            int idx = tid + i * 256;
            int r   = idx >> 5;
            int c   = bc[i];
            Bs[c + 0][r] = (f16)bv[i][0];
            Bs[c + 1][r] = (f16)bv[i][1];
            Bs[c + 2][r] = (f16)bv[i][2];
            Bs[c + 3][r] = (f16)bv[i][3];
        }
        __syncthreads();

        // ---- fragments (ISA 7.12.2 16-bit layouts) + 4 WMMAs per wave ----
        v16h af;
        const int am = wm * 16 + l16;
        #pragma unroll
        for (int r = 0; r < 8; ++r) {
            // A: VGPR r holds K = 16*(r/4) + 2*(r%4) + 8*half (+0,+1)
            int kk = ((r >> 2) << 4) + ((r & 3) << 1) + (hsel << 3);
            af[2 * r]     = As[am][kk];
            af[2 * r + 1] = As[am][kk + 1];
        }
        #pragma unroll
        for (int j = 0; j < 4; ++j) {
            v16h bf;
            const int bn = wn * 64 + j * 16 + l16;
            #pragma unroll
            for (int r = 0; r < 8; ++r) {
                // B: VGPR r holds K = 16*(r/4) + 2*(r%4) + 8*half (+0,+1), N = lane&15
                int kk = ((r >> 2) << 4) + ((r & 3) << 1) + (hsel << 3);
                bf[2 * r]     = Bs[bn][kk];
                bf[2 * r + 1] = Bs[bn][kk + 1];
            }
            acc[j] = wmma_f16(af, bf, acc[j]);
        }
        __syncthreads();
    }

    // ---- store: C/D layout — VGPR r is row (r + 8*half), col = lane&15 ----
    #pragma unroll
    for (int j = 0; j < 4; ++j) {
        #pragma unroll
        for (int r = 0; r < 8; ++r) {
            int gm = m0 + wm * 16 + (hsel << 3) + r;
            int gn = n0 + wn * 64 + j * 16 + l16;
            if (gn < N)
                C[(size_t)gm * ldc + gn] = acc[j][r];
        }
    }
}

// ---------------------------------------------------------------------------
// x = condition + pe (pe broadcast over batch), float4 vectorized
// ---------------------------------------------------------------------------
__global__ __launch_bounds__(256)
void add_pe_kernel(const float* __restrict__ cond, const float* __restrict__ pe,
                   float* __restrict__ x, int total4, int LD4)
{
    int i = blockIdx.x * blockDim.x + threadIdx.x;
    if (i >= total4) return;
    v4f c = ((const v4f*)cond)[i];
    v4f p = ((const v4f*)pe)[i % LD4];
    ((v4f*)x)[i] = c + p;
}

// ---------------------------------------------------------------------------
// Causal depthwise conv (K=4) over L within each batch + SiLU.
// Input xin = xz[:, :DI] (stride twoDI).  Output xc (stride DI).
// ---------------------------------------------------------------------------
__global__ __launch_bounds__(256)
void conv_silu_kernel(const float* __restrict__ xz, const float* __restrict__ cw,
                      const float* __restrict__ cb, float* __restrict__ xc,
                      int DI, int twoDI)
{
    int i = blockIdx.x * blockDim.x + threadIdx.x;
    int total = MROWS * DI;
    if (i >= total) return;
    int d  = i % DI;
    int bl = i / DI;            // b*L + l
    int l  = bl % SEQL;
    float acc = cb[d];
    #pragma unroll
    for (int k = 0; k < KCONV; ++k) {
        int ll = l + k - (KCONV - 1);
        float w = cw[d * KCONV + k];
        float v = (ll >= 0) ? xz[(size_t)(bl + k - (KCONV - 1)) * twoDI + d] : 0.0f;
        acc += w * v;
    }
    xc[i] = acc / (1.0f + __expf(-acc));    // SiLU
}

// ---------------------------------------------------------------------------
// dt = softplus(dtproj + b_dt)   (in-place on dtproj)
// ---------------------------------------------------------------------------
__global__ __launch_bounds__(256)
void softplus_bias_kernel(float* __restrict__ dt, const float* __restrict__ bdt,
                          int total, int DI)
{
    int i = blockIdx.x * blockDim.x + threadIdx.x;
    if (i >= total) return;
    float v = dt[i] + bdt[i % DI];
    dt[i] = (v > 20.0f) ? v : log1pf(__expf(v));
}

// ---------------------------------------------------------------------------
// Selective-scan: one thread per (b, d) channel, h[64] state in VGPRs.
// Fused epilogue: y = (scan + xc*Dskip) * silu(z).
// Grid: (DI/128, B), 128 threads.
// ---------------------------------------------------------------------------
__global__ __launch_bounds__(128)
void scan_kernel(const float* __restrict__ dt, const float* __restrict__ proj,
                 const float* __restrict__ xc, const float* __restrict__ xz,
                 const float* __restrict__ A_log, const float* __restrict__ Dskip,
                 float* __restrict__ y,
                 int DI, int projld, int dtRank, int twoDI)
{
    __shared__ float sB[NSTATE];
    __shared__ float sC[NSTATE];

    const int b = blockIdx.y;
    const int d = blockIdx.x * 128 + threadIdx.x;
    const int t = threadIdx.x;

    float negA[NSTATE], h[NSTATE];
    #pragma unroll
    for (int n = 0; n < NSTATE; ++n) {
        negA[n] = -__expf(A_log[(size_t)d * NSTATE + n]);
        h[n] = 0.0f;
    }
    const float dsk = Dskip[d];

    for (int l = 0; l < SEQL; ++l) {
        const size_t row = (size_t)(b * SEQL + l);
        if (t < NSTATE)        sB[t]          = proj[row * projld + dtRank + t];
        else if (t < 2*NSTATE) sC[t - NSTATE] = proj[row * projld + dtRank + NSTATE + (t - NSTATE)];
        __syncthreads();

        const float dtv = dt[row * DI + d];
        const float xcv = xc[row * DI + d];
        const float dbx = dtv * xcv;
        float acc = 0.0f;
        #pragma unroll
        for (int n = 0; n < NSTATE; ++n) {
            h[n] = __expf(dtv * negA[n]) * h[n] + dbx * sB[n];
            acc += h[n] * sC[n];
        }
        const float zv = xz[row * twoDI + DI + d];
        const float g  = zv / (1.0f + __expf(-zv));     // silu(z)
        y[row * DI + d] = (acc + xcv * dsk) * g;
        __syncthreads();
    }
}

// ---------------------------------------------------------------------------
// Host-side orchestration
// ---------------------------------------------------------------------------
static inline void launch_gemm(const float* A, int lda, const float* W, int ldb,
                               float* C, int ldc, int N, int K, hipStream_t s)
{
    dim3 grid((N + BN - 1) / BN, MROWS / BM);
    gemm_wmma_f16_kernel<<<grid, 256, 0, s>>>(A, lda, W, ldb, C, ldc, N, K);
}

extern "C" void kernel_launch(void* const* d_in, const int* in_sizes, int n_in,
                              void* d_out, int out_size, void* d_ws, size_t ws_size,
                              hipStream_t stream)
{
    (void)in_sizes; (void)n_in; (void)out_size; (void)ws_size;

    const float* cond   = (const float*)d_in[0];
    const float* pe     = (const float*)d_in[2];
    const float* W_in1  = (const float*)d_in[3];
    const float* cw1    = (const float*)d_in[4];
    const float* cb1    = (const float*)d_in[5];
    const float* W_x1   = (const float*)d_in[6];
    const float* W_dt1  = (const float*)d_in[7];
    const float* bdt1   = (const float*)d_in[8];
    const float* Alog1  = (const float*)d_in[9];
    const float* D1p    = (const float*)d_in[10];
    const float* W_out1 = (const float*)d_in[11];
    const float* W_in2  = (const float*)d_in[12];
    const float* cw2    = (const float*)d_in[13];
    const float* cb2    = (const float*)d_in[14];
    const float* W_x2   = (const float*)d_in[15];
    const float* W_dt2  = (const float*)d_in[16];
    const float* bdt2   = (const float*)d_in[17];
    const float* Alog2  = (const float*)d_in[18];
    const float* D2p    = (const float*)d_in[19];
    const float* W_out2 = (const float*)d_in[20];
    float* out = (float*)d_out;

    const int D1 = 1024, DI1 = 2048, R1 = 64,  P1 = R1 + 2 * NSTATE;  // 192
    const int DI2 = 4096, D2 = 2048, R2 = 128, P2 = R2 + 2 * NSTATE;  // 256
    const int M = MROWS;

    // ---- workspace arena (block1 and block2 regions overlap; serial stream) ----
    float* ws    = (float*)d_ws;
    float* x2    = ws;                                   // M*D1
    float* arena = ws + (size_t)M * D1;
    // block1 layout
    float* x1    = arena;                                // M*D1
    float* xz1   = x1    + (size_t)M * D1;               // M*2*DI1
    float* xc1   = xz1   + (size_t)M * 2 * DI1;          // M*DI1
    float* proj1 = xc1   + (size_t)M * DI1;              // M*P1
    float* dt1   = proj1 + (size_t)M * P1;               // M*DI1
    float* y1    = dt1   + (size_t)M * DI1;              // M*DI1
    // block2 layout (reuses arena; block1 buffers dead by then)
    float* xz2   = arena;                                // M*2*DI2
    float* xc2   = xz2   + (size_t)M * 2 * DI2;          // M*DI2
    float* proj2 = xc2   + (size_t)M * DI2;              // M*P2
    float* dt2   = proj2 + (size_t)M * P2;               // M*DI2
    float* y2    = dt2   + (size_t)M * DI2;              // M*DI2

    // ================= Block 1 =================
    {
        int total4 = (M * D1) / 4;
        add_pe_kernel<<<(total4 + 255) / 256, 256, 0, stream>>>(cond, pe, x1, total4, (SEQL * D1) / 4);
    }
    launch_gemm(x1, D1, W_in1, 2 * DI1, xz1, 2 * DI1, 2 * DI1, D1, stream);
    {
        int total = M * DI1;
        conv_silu_kernel<<<(total + 255) / 256, 256, 0, stream>>>(xz1, cw1, cb1, xc1, DI1, 2 * DI1);
    }
    launch_gemm(xc1, DI1, W_x1, P1, proj1, P1, P1, DI1, stream);
    launch_gemm(proj1, P1, W_dt1, DI1, dt1, DI1, DI1, R1, stream);
    {
        int total = M * DI1;
        softplus_bias_kernel<<<(total + 255) / 256, 256, 0, stream>>>(dt1, bdt1, total, DI1);
    }
    {
        dim3 grid(DI1 / 128, BATCH);
        scan_kernel<<<grid, 128, 0, stream>>>(dt1, proj1, xc1, xz1, Alog1, D1p, y1,
                                              DI1, P1, R1, 2 * DI1);
    }
    launch_gemm(y1, DI1, W_out1, D1, x2, D1, D1, DI1, stream);

    // ================= Block 2 =================
    launch_gemm(x2, D1, W_in2, 2 * DI2, xz2, 2 * DI2, 2 * DI2, D1, stream);
    {
        int total = M * DI2;
        conv_silu_kernel<<<(total + 255) / 256, 256, 0, stream>>>(xz2, cw2, cb2, xc2, DI2, 2 * DI2);
    }
    launch_gemm(xc2, DI2, W_x2, P2, proj2, P2, P2, DI2, stream);
    launch_gemm(proj2, P2, W_dt2, DI2, dt2, DI2, DI2, R2, stream);
    {
        int total = M * DI2;
        softplus_bias_kernel<<<(total + 255) / 256, 256, 0, stream>>>(dt2, bdt2, total, DI2);
    }
    {
        dim3 grid(DI2 / 128, BATCH);
        scan_kernel<<<grid, 128, 0, stream>>>(dt2, proj2, xc2, xz2, Alog2, D2p, y2,
                                              DI2, P2, R2, 2 * DI2);
    }
    launch_gemm(y2, DI2, W_out2, D2, out, D2, D2, DI2, stream);
}
